// GNN_1331439862190
// MI455X (gfx1250) — compile-verified, matrix operand order checked by reading, not run on hardware
//
#include <hip/hip_runtime.h>

// GNN GRU cell (SR-GNN) for MI455X / gfx1250.
// Launch 1: prep kernel pre-converts + pre-swizzles all GEMM weights into d_ws
//           (bf16, exact WMMA B-fragment lane layout; needs 352KB of d_ws).
// Launch 2: one workgroup (8 wave32) per batch row; all GEMMs via
//           v_wmma_f32_16x16x32_bf16 (bf16 in, f32 accum); activations staged
//           in LDS (136KB/WG; CDNA5 WGP has 320KB LDS). The f32 hidden copy is
//           staged with global_load_async_to_lds_b128 (ASYNCcnt path).

typedef __attribute__((ext_vector_type(16))) __bf16 v16bf;
typedef __attribute__((ext_vector_type(8)))  float  v8f;
typedef __attribute__((ext_vector_type(4)))  float  v4f;
typedef __attribute__((ext_vector_type(2)))  float  v2f;
typedef __attribute__((ext_vector_type(4)))  __bf16 v4bf;
typedef __attribute__((ext_vector_type(2)))  __bf16 v2bf;

#define B_N   4096
#define L_N   50
#define E_N   128
#define LP    64          // L padded to multiple of 16 (and of K=32 for GEMM2)

// LDS row strides (elements), padded to dodge 64-bank alignment
#define SA_S   136        // A_cat   bf16 [64][128] : cols 0..63 A_in, 64..127 A_out
#define SC_S   392        // comb    bf16 [64][384] : 0..255 inputs, 256..383 hidden
#define SHT_S  136        // hT      bf16 [128][128]: cols 0..63 h_inT, 64..127 h_outT
#define SH_S   132        // hidden  f32  [64][128] (for epilogue blend; rows >=50 unused)

#define SA_BYTES   (LP  * SA_S  * 2)
#define SC_BYTES   (LP  * SC_S  * 2)
#define SHT_BYTES  (E_N * SHT_S * 2)
#define SH_BYTES   (LP  * SH_S  * 4)
#define SMEM_BYTES (SA_BYTES + SC_BYTES + SHT_BYTES + SH_BYTES)   // 136192 B

// Swizzled weight workspace layout (bf16 elements):
//   GEMM1 ([w_in;w_out], 16 N-tiles x 4 k-steps)  : blocks [t*4+kk],  64 KB
//   GEMM3 ([w_ih|w_hh],  24 N-tiles x 12 k-steps) : blocks [tt*12+kk], 288 KB
// Each block: 32 lanes x 16 bf16 (two 16B chunks per lane) = 1 KB, wave-contiguous.
#define WS_G1_BLOCKS  (16 * 4)
#define WS_G3_BLOCKS  (24 * 12)
#define WS_BLOCKS     (WS_G1_BLOCKS + WS_G3_BLOCKS)      // 352
#define WS_G3_OFF     (WS_G1_BLOCKS * 32 * 16)           // bf16 elements
#define WS_BYTES      (WS_BLOCKS * 32 * 16 * 2)          // 360448 B

// ---------------- weight pre-swizzle kernel ----------------
// One thread = one lane of one fragment block: writes 32B of bf16.
// Fragment layout (ISA 7.12.2, 16-bit 16x32): lane&15 = B^T row (= weight row),
// K chunks {lo..lo+7} and {lo+16..lo+23}, lo = kbase + (lane>=16 ? 8 : 0).
__global__ void swizzle_weights_kernel(const float* __restrict__ w_in,
                                       const float* __restrict__ w_out,
                                       const float* __restrict__ w_ih,
                                       const float* __restrict__ w_hh,
                                       __bf16* __restrict__ ws) {
  int idx  = blockIdx.x * 256 + threadIdx.x;
  int blk  = idx >> 5;
  if (blk >= WS_BLOCKS) return;
  int lane = idx & 31;
  int l16  = lane & 15;
  int klo  = (lane >> 4) * 8;
  const float* src;
  int kb;
  if (blk < WS_G1_BLOCKS) {                  // GEMM1: t 0..15, kk 0..3
    int t = blk >> 2, kk = blk & 3;
    src = (t < 8 ? w_in : w_out) + (size_t)(16 * (t & 7) + l16) * E_N;
    kb  = kk * 32 + klo;
  } else {                                   // GEMM3: tt 0..23, kk 0..11
    int b2 = blk - WS_G1_BLOCKS;
    int tt = b2 / 12, kk = b2 % 12;
    int j  = 16 * tt + l16;                  // Wcat row 0..383
    kb = kk * 32 + klo;
    if (kb < 256) { src = w_ih + (size_t)j * 256; }
    else          { src = w_hh + (size_t)j * 128; kb -= 256; }
  }
  __bf16* dst = ws + (size_t)idx * 16;
#pragma unroll
  for (int i = 0; i < 8; ++i) {
    dst[i]     = (__bf16)src[kb + i];
    dst[8 + i] = (__bf16)src[kb + 16 + i];
  }
}

// ---------------- fragment helpers ----------------
__device__ __forceinline__ v16bf frag_lds(const __bf16* p, int lo) {
  union { v16bf v; uint4 q[2]; } u;
  u.q[0] = *reinterpret_cast<const uint4*>(p + lo);
  u.q[1] = *reinterpret_cast<const uint4*>(p + lo + 16);
  return u.v;
}

__device__ __forceinline__ v16bf frag_ws(const __bf16* __restrict__ base,
                                         int blkLin, int lane) {
  const uint4* p =
      reinterpret_cast<const uint4*>(base + ((size_t)blkLin * 32 + lane) * 16);
  union { v16bf v; uint4 q[2]; } u;
  u.q[0] = p[0];
  u.q[1] = p[1];
  return u.v;
}

__device__ __forceinline__ v8f wmma_bf(v16bf a, v16bf b, v8f c) {
  return __builtin_amdgcn_wmma_f32_16x16x32_bf16(
      /*neg_a=*/false, a, /*neg_b=*/false, b,
      /*c_mod=*/(short)0, c, /*reuse_a=*/false, /*reuse_b=*/false);
}

// Async global->LDS copy of 16B per active lane (GV mode), tracked by ASYNCcnt.
__device__ __forceinline__ void async_load_b128(unsigned int lds_byte_off,
                                                const void* gptr) {
  asm volatile("global_load_async_to_lds_b128 %0, %1, off"
               :: "v"(lds_byte_off), "v"(gptr) : "memory");
}

// ---------------- main fused kernel ----------------
__global__ __launch_bounds__(256, 1) void gnn_cell_kernel(
    const float* __restrict__ A,     const float* __restrict__ hidden,
    const float* __restrict__ b_ih,  const float* __restrict__ b_hh,
    const float* __restrict__ b_iah, const float* __restrict__ b_ioh,
    const float* __restrict__ b_in,  const float* __restrict__ b_out,
    const __bf16* __restrict__ ws,   float* __restrict__ out) {
  extern __shared__ char smem[];
  __bf16* sA    = (__bf16*)(smem);
  __bf16* sComb = (__bf16*)(smem + SA_BYTES);
  __bf16* sHT   = (__bf16*)(smem + SA_BYTES + SC_BYTES);
  float*  sH    = (float*)(smem + SA_BYTES + SC_BYTES + SHT_BYTES);

  const int b    = blockIdx.x;
  const int tid  = threadIdx.x;
  const int wv   = tid >> 5;
  const int lane = tid & 31;
  const int l16  = lane & 15;
  const int hiL  = lane >> 4;                 // 0 / 1
  const int m    = wv & 3;                    // M tile (rows 16m..16m+15)
  const int half = wv >> 2;                   // 0: "in"/low-N half, 1: "out"/high-N
  const int mrow = 16 * m + l16;              // A-fragment row for this lane
  const int klo  = hiL * 8;                   // per-lane K offset

  const v8f vzero = {0.f, 0.f, 0.f, 0.f, 0.f, 0.f, 0.f, 0.f};

  const float* Ab = A      + (size_t)b * L_N * (2 * L_N);
  const float* Hb = hidden + (size_t)b * L_N * E_N;

  // ---------- stage 0a: kick off async f32 hidden -> sH (16B per lane) ----------
  {
    unsigned int sHb = __builtin_amdgcn_groupstaticsize() +
                       (unsigned int)(SA_BYTES + SC_BYTES + SHT_BYTES);
    for (int idx = tid; idx < L_N * (E_N / 4); idx += 256) {
      int r = idx >> 5, q = idx & 31;         // row, float4-chunk
      async_load_b128(sHb + (unsigned int)(r * SH_S + 4 * q) * 4,
                      Hb + r * E_N + 4 * q);
    }
  }

  // ---------- stage 0b: stage A (bf16, split 50|50 -> 64|64, padded) ----------
  for (int idx = tid; idx < L_N * L_N; idx += 256) {   // 2500 float2 chunks
    int r = idx / L_N, p = idx % L_N;
    v2f v = *reinterpret_cast<const v2f*>(Ab + r * (2 * L_N) + 2 * p);
    int c  = 2 * p;
    int dc = (c < L_N) ? c : (64 + c - L_N);
    *reinterpret_cast<v2bf*>(&sA[r * SA_S + dc]) = __builtin_convertvector(v, v2bf);
  }
  // zero pad: rows 0..49, col-pairs 25..31 (cols 50..63) and 57..63 (114..127)
  for (int idx = tid; idx < L_N * 14; idx += 256) {
    int r = idx / 14, k = idx % 14;
    int cp = (k < 7) ? (25 + k) : (50 + k);
    *reinterpret_cast<unsigned int*>(&sA[r * SA_S + 2 * cp]) = 0u;
  }
  // zero pad: rows 50..63 of sA (full 136-element rows, as u32)
  for (int idx = tid; idx < 14 * 68; idx += 256) {
    int r = L_N + idx / 68, cp = idx % 68;
    *reinterpret_cast<unsigned int*>(&sA[r * SA_S + 2 * cp]) = 0u;
  }
  // zero pad: sComb hidden region rows 50..63 (cols 256..383, as u32)
  for (int idx = tid; idx < 14 * 64; idx += 256) {
    int r = L_N + idx / 64, cp = idx % 64;
    *reinterpret_cast<unsigned int*>(&sComb[r * SC_S + 256 + 2 * cp]) = 0u;
  }
  asm volatile("s_wait_asynccnt 0" ::: "memory");
  __syncthreads();

  // ---------- stage 0c: bf16 hidden from sH (f32 LDS) into sComb[:,256:384] ----------
  for (int idx = tid; idx < L_N * (E_N / 4); idx += 256) {
    int r = idx >> 5, q = idx & 31;
    v4f v = *reinterpret_cast<const v4f*>(&sH[r * SH_S + 4 * q]);
    *reinterpret_cast<v4bf*>(&sComb[r * SC_S + 256 + 4 * q]) =
        __builtin_convertvector(v, v4bf);
  }
  __syncthreads();

  // ---------- GEMM1: [h_in | h_out] = hidden @ [w_in ; w_out]^T, store transposed ----------
  {
    v8f acc[8];
#pragma unroll
    for (int t = 0; t < 8; ++t) acc[t] = vzero;
    for (int kk = 0; kk < 4; ++kk) {          // K = 128
      v16bf a = frag_lds(sComb + mrow * SC_S + 256, kk * 32 + klo);
#pragma unroll
      for (int t = 0; t < 8; ++t) {
        v16bf bf = frag_ws(ws, (8 * half + t) * 4 + kk, lane);
        acc[t] = wmma_bf(a, bf, acc[t]);
      }
    }
    const float* bias = half ? b_out : b_in;
    const int colbase = 64 * half + 16 * m + 8 * hiL;   // transposed col base
#pragma unroll
    for (int t = 0; t < 8; ++t) {
      int j = 16 * t + l16;                   // feature index 0..127
      float bv = bias[j];
#pragma unroll
      for (int r = 0; r < 8; r += 2) {        // pack row-pairs: adjacent cols
        v2f v = {acc[t][r] + bv, acc[t][r + 1] + bv};
        *reinterpret_cast<v2bf*>(&sHT[j * SHT_S + colbase + r]) =
            __builtin_convertvector(v, v2bf);
      }
    }
  }
  __syncthreads();

  // ---------- GEMM2: input_in/out = A_in/out @ h_in/out + b_iah/b_ioh (K=64) ----------
  {
    v8f acc[8];
#pragma unroll
    for (int t = 0; t < 8; ++t) acc[t] = vzero;
    const int coff = 64 * half;               // column offset into sA / sHT
    for (int kk = 0; kk < 2; ++kk) {
      int kb = kk * 32;
      v16bf a = frag_lds(sA + mrow * SA_S + coff, kb + klo);
#pragma unroll
      for (int t = 0; t < 8; ++t) {
        int j = 16 * t + l16;                 // feature column 0..127
        v16bf bf = frag_lds(sHT + j * SHT_S + coff, kb + klo);
        acc[t] = wmma_bf(a, bf, acc[t]);
      }
    }
    const float* bias = half ? b_ioh : b_iah;
#pragma unroll
    for (int t = 0; t < 8; ++t) {
      int j = 16 * t + l16;
      float bv = bias[j];
#pragma unroll
      for (int r = 0; r < 8; ++r) {
        int l = 16 * m + r + 8 * hiL;
        sComb[l * SC_S + 128 * half + j] = (__bf16)(acc[t][r] + bv);
      }
    }
  }
  __syncthreads();

  // ---------- GEMM3: gates. comb=[inputs|hidden] [64,384], Wcat=[w_ih|w_hh].
  // r/i gates: single K=384 accumulation of gi+gh. n gate: split K at 256 so
  // i_n and h_n stay separate (reset gate multiplies h_n only).
  // Two passes of 2 column-tiles each to keep live accumulators at 8 v8f. ----------
  {
    const __bf16* wsG3 = ws + WS_G3_OFF;
    const int cb = 4 * half;                  // this wave's N-tile base (0 or 4)
    float* Ob = out + (size_t)b * L_N * E_N;
    for (int pass = 0; pass < 2; ++pass) {
      v8f aR[2], aI[2], aN[2], aH[2];
#pragma unroll
      for (int u = 0; u < 2; ++u) { aR[u] = vzero; aI[u] = vzero;
                                    aN[u] = vzero; aH[u] = vzero; }
      for (int kk = 0; kk < 12; ++kk) {       // K = 384
        v16bf a = frag_lds(sComb + mrow * SC_S, kk * 32 + klo);
        bool isH = (kk >= 8);                 // uniform: hidden part of K
#pragma unroll
        for (int u = 0; u < 2; ++u) {
          int ct = cb + 2 * pass + u;
          v16bf bR = frag_ws(wsG3, ct * 12 + kk, lane);
          v16bf bI = frag_ws(wsG3, (ct + 8) * 12 + kk, lane);
          v16bf bN = frag_ws(wsG3, (ct + 16) * 12 + kk, lane);
          aR[u] = wmma_bf(a, bR, aR[u]);
          aI[u] = wmma_bf(a, bI, aI[u]);
          if (!isH) aN[u] = wmma_bf(a, bN, aN[u]);
          else      aH[u] = wmma_bf(a, bN, aH[u]);
        }
      }
      // In-register GRU epilogue; C/D layout: elem(M = r + 8*(lane>>4), N = lane&15)
#pragma unroll
      for (int u = 0; u < 2; ++u) {
        int e = 16 * (cb + 2 * pass + u) + l16;   // output feature 0..127
        float br  = b_ih[e]       + b_hh[e];
        float bi  = b_ih[e + 128] + b_hh[e + 128];
        float bnI = b_ih[e + 256];
        float bnH = b_hh[e + 256];
#pragma unroll
        for (int r = 0; r < 8; ++r) {
          int l = 16 * m + r + 8 * hiL;
          if (l < L_N) {
            float rg = 1.f / (1.f + __expf(-(aR[u][r] + br)));
            float ig = 1.f / (1.f + __expf(-(aI[u][r] + bi)));
            float ng = tanhf((aN[u][r] + bnI) + rg * (aH[u][r] + bnH));
            float h  = sH[l * SH_S + e];
            Ob[l * E_N + e] = (1.f - ig) * h + ig * ng;
          }
        }
      }
    }
  }
}

extern "C" void kernel_launch(void* const* d_in, const int* in_sizes, int n_in,
                              void* d_out, int out_size, void* d_ws, size_t ws_size,
                              hipStream_t stream) {
  (void)in_sizes; (void)n_in; (void)out_size; (void)ws_size;  // needs ws >= 352KB
  const float* A     = (const float*)d_in[0];
  const float* hidden= (const float*)d_in[1];
  const float* w_ih  = (const float*)d_in[2];
  const float* w_hh  = (const float*)d_in[3];
  const float* b_ih  = (const float*)d_in[4];
  const float* b_hh  = (const float*)d_in[5];
  const float* b_iah = (const float*)d_in[6];
  const float* b_ioh = (const float*)d_in[7];
  const float* w_in  = (const float*)d_in[8];
  const float* b_in  = (const float*)d_in[9];
  const float* w_out = (const float*)d_in[10];
  const float* b_out = (const float*)d_in[11];
  __bf16* ws = (__bf16*)d_ws;

  int prep_threads = WS_BLOCKS * 32;
  swizzle_weights_kernel<<<(prep_threads + 255) / 256, 256, 0, stream>>>(
      w_in, w_out, w_ih, w_hh, ws);
  gnn_cell_kernel<<<B_N, 256, SMEM_BYTES, stream>>>(
      A, hidden, b_ih, b_hh, b_iah, b_ioh, b_in, b_out, ws, (float*)d_out);
}
